// RelativeMultiHeadAttention_22376779612200
// MI455X (gfx1250) — matrix operand order repaired and checked by reading, not compiled, not run on hardware
//
#include <hip/hip_runtime.h>

// ---------------------------------------------------------------------------
// CDNA5 (gfx1250) WMMA implementation of relative multi-head attention.
// bf16 WMMA (v_wmma_f32_16x16x32_bf16), fp32 accumulate, flash-style fusion,
// TDM (tensor_load_to_lds) + async global->LDS staging where available.
// ---------------------------------------------------------------------------

typedef __attribute__((ext_vector_type(16))) __bf16 v16bf;
typedef __attribute__((ext_vector_type(8)))  float  v8f;
typedef __attribute__((ext_vector_type(4)))  unsigned int v4u;
typedef __attribute__((ext_vector_type(8)))  int v8i;
typedef __attribute__((ext_vector_type(4)))  int v4i;
typedef int v4i_b __attribute__((vector_size(16)));   // gcc-style, matches builtin proto

#define D_MODEL   1024
#define NUM_HEADS 16
#define DK        64
#define MAX_LEN   2048
#define BATCH     2
#define SEQ       2048
#define MTOT      (BATCH * SEQ)      // 4096 rows
#define RELROWS   (2 * MAX_LEN)      // 4096 rel_emb rows

#define AS1 __attribute__((address_space(1)))
#define AS3 __attribute__((address_space(3)))

// CDNA5 feature gates (device pass only; safe fallbacks otherwise)
#if defined(__HIP_DEVICE_COMPILE__) && defined(__gfx1250__) && \
    __has_builtin(__builtin_amdgcn_global_load_async_to_lds_b128) && \
    __has_builtin(__builtin_amdgcn_s_wait_asynccnt)
#define USE_ASYNC_LDS 1
#else
#define USE_ASYNC_LDS 0
#endif

#if defined(__HIP_DEVICE_COMPILE__) && defined(__gfx1250__) && \
    __has_builtin(__builtin_amdgcn_tensor_load_to_lds) && \
    __has_builtin(__builtin_amdgcn_s_wait_tensorcnt)
#define USE_TDM 1
#else
#define USE_TDM 0
#endif

#if __has_include(<hip/amd_detail/amd_gfx1250_TDM.h>)
#define TDM_SIX_ARG 1
#else
#define TDM_SIX_ARG 0
#endif

// ---- fp32 -> bf16 (round-to-nearest-even) ---------------------------------
static __device__ __forceinline__ __bf16 f2bf(float f) {
    unsigned u; __builtin_memcpy(&u, &f, 4);
    unsigned r = (u + 0x7FFFu + ((u >> 16) & 1u)) >> 16;
    unsigned short s = (unsigned short)r;
    __bf16 b; __builtin_memcpy(&b, &s, 2);
    return b;
}

static __device__ __forceinline__ v8f zero8() {
    v8f z;
#pragma unroll
    for (int i = 0; i < 8; ++i) z[i] = 0.0f;
    return z;
}

// 16 contiguous bf16 (32B) -> one fragment register file
static __device__ __forceinline__ v16bf load16bf(const __bf16* p) {
    p = (const __bf16*)__builtin_assume_aligned(p, 32);
    v16bf r; __builtin_memcpy(&r, p, 32);
    return r;
}
// two 8-element (16B) chunks -> one fragment (A-matrix K split layout)
static __device__ __forceinline__ v16bf load8x2bf(const __bf16* p0, const __bf16* p1) {
    p0 = (const __bf16*)__builtin_assume_aligned(p0, 16);
    p1 = (const __bf16*)__builtin_assume_aligned(p1, 16);
    v16bf r;
    __builtin_memcpy(&r, p0, 16);
    __builtin_memcpy((char*)&r + 16, p1, 16);
    return r;
}

// ---------------------------------------------------------------------------
// Kernel 1: fp32 -> bf16 conversion
// ---------------------------------------------------------------------------
__global__ void cvt_f32_bf16(const float* __restrict__ src,
                             __bf16* __restrict__ dst, int n) {
    int i = blockIdx.x * blockDim.x + threadIdx.x;
    if (i < n) dst[i] = f2bf(src[i]);
}

// ---------------------------------------------------------------------------
// Kernel 2: C = A @ W^T + bias    (A: [MTOT x K] bf16, W: [N x K] bf16)
// One wave computes one 16x16 tile, K/32 WMMAs.
// mode 0: bf16 head-split [B,H,L,dk];  mode 1: fp32 [MTOT x D_MODEL]
// ---------------------------------------------------------------------------
__global__ __launch_bounds__(32) void gemm_wmma_bf16(
    const __bf16* __restrict__ A,
    const __bf16* __restrict__ W,
    const float*  __restrict__ bias,
    __bf16* __restrict__ outHeads,
    float*  __restrict__ outPlain,
    int K, int mode)
{
    const int m0   = blockIdx.x * 16;
    const int n0   = blockIdx.y * 16;
    const int lane = threadIdx.x & 31;
    const int lr   = lane & 15;
    const int half = lane >> 4;

    const __bf16* Ar = A + (size_t)(m0 + lr) * K;
    const __bf16* Wr = W + (size_t)(n0 + lr) * K;

    v8f acc = zero8();
    for (int k0 = 0; k0 < K; k0 += 32) {
        // A layout: lanes 0-15 -> K {k0+0..7, k0+16..23}; lanes 16-31 -> {+8..15, +24..31}
        v16bf a = load8x2bf(Ar + k0 + half * 8, Ar + k0 + 16 + half * 8);
        // B layout: same K split as A: lanes 0-15 -> {k0+0..7, k0+16..23}; lanes 16-31 -> {+8..15, +24..31}
        v16bf b = load8x2bf(Wr + k0 + half * 8, Wr + k0 + 16 + half * 8);
        acc = __builtin_amdgcn_wmma_f32_16x16x32_bf16(false, a, false, b,
                                                      (short)0, acc, false, false);
    }

    const int n  = n0 + lr;
    const float bn = bias[n];
#pragma unroll
    for (int r = 0; r < 8; ++r) {
        const int m = m0 + r + 8 * half;       // C layout: row = r + 8*half
        const float v = acc[r] + bn;
        if (mode == 0) {
            const int bidx = m >> 11;          // / SEQ
            const int li   = m & (SEQ - 1);
            const int h    = n >> 6;           // / DK
            const int d    = n & (DK - 1);
            outHeads[((size_t)(bidx * NUM_HEADS + h) * SEQ + li) * DK + d] = f2bf(v);
        } else {
            outPlain[(size_t)m * D_MODEL + n] = v;
        }
    }
}

// ---------------------------------------------------------------------------
// Kernel 3: fused relative attention (flash-style, online softmax).
// Grid: (SEQ/64, H, B); block = 128 threads = 4 waves; wave w owns a 16-row
// i-tile. Per 16-wide j-tile:
//   K tile  -> LDS via TDM tensor_load_to_lds (fallback: vector copy)
//   V tile  -> LDS via global_load_async_to_lds_b128, then LDS transpose
//   content : 2 WMMAs   positional rel-window: 4 WMMAs + LDS shear gather
//   P @ V   : 4 WMMAs (K padded 16->32)
// ---------------------------------------------------------------------------
__global__ __launch_bounds__(128) void rel_attn_wmma(
    const __bf16* __restrict__ Qh,   // [B,H,L,dk]
    const __bf16* __restrict__ Kh,
    const __bf16* __restrict__ Vh,
    const __bf16* __restrict__ Rel,  // [2*MAX_LEN, dk]
    __bf16* __restrict__ AO)         // [B*L, D_MODEL], col = h*dk + d
{
    alignas(32) __shared__ __bf16 Ksh[16][DK];     // 2 KB  (row-major K tile)
    alignas(32) __shared__ __bf16 Vraw[16][DK];    // 2 KB  (row-major V tile)
    alignas(32) __shared__ __bf16 VshT[DK][16];    // 2 KB  (transposed V tile)
    alignas(32) __shared__ float  Tsh[4][16][32];  // 8 KB  (per-wave pos strips)
    alignas(32) __shared__ __bf16 Psh[4][16][16];  // 2 KB  (per-wave P round-trip)

    const int w    = threadIdx.x >> 5;
    const int lane = threadIdx.x & 31;
    const int lr   = lane & 15;
    const int half = lane >> 4;
    const int h    = blockIdx.y;
    const int b    = blockIdx.z;
    const int i0   = blockIdx.x * 64 + w * 16;

    const size_t bh = (size_t)(b * NUM_HEADS + h) * SEQ;
    const __bf16* Qrow = Qh + (bh + i0 + lr) * DK;

    // Q A-fragments, resident for the whole j loop (K split 0..31 / 32..63)
    v16bf aq0 = load16bf(Qrow + half * 16);
    v16bf aq1 = load16bf(Qrow + 32 + half * 16);

    float mrow[8], ssum[8];
#pragma unroll
    for (int r = 0; r < 8; ++r) { mrow[r] = -1e30f; ssum[r] = 0.0f; }
    v8f o0 = zero8(), o1 = zero8(), o2 = zero8(), o3 = zero8();

    const float scale = 0.125f;            // 1/sqrt(dk)
    const __bf16 zb = f2bf(0.0f);

    // staging assignment: 128 threads -> 16 rows x (8 threads x 8 elems)
    const int srow = threadIdx.x >> 3;
    const int scol = (threadIdx.x & 7) * 8;

    for (int j0 = 0; j0 < SEQ; j0 += 16) {
        __syncthreads();                   // previous tile fully consumed
        const __bf16* kp = Kh + (bh + j0 + srow) * DK + scol;
        const __bf16* vp = Vh + (bh + j0 + srow) * DK + scol;

#if USE_TDM
        if (w == 0) {
            // D# for a 16x64 bf16 tile (row stride 64 elems) -> Ksh
            const unsigned long long ga =
                (unsigned long long)(size_t)(Kh + (bh + (size_t)j0) * DK);
            const unsigned ldsa =
                (unsigned)(unsigned long long)(AS3 const void*)&Ksh[0][0];
            v4u g0; v8i g1; v4i g2, g3;
            g0[0] = 1u;                                       // count=1, user desc
            g0[1] = ldsa;                                     // lds_addr
            g0[2] = (unsigned)(ga & 0xFFFFFFFFu);             // global_addr[31:0]
            g0[3] = (unsigned)((ga >> 32) & 0x01FFFFFFu) | (2u << 30); // addr[56:32]|type=2
            g1[0] = (int)(1u << 16);                          // data_size=2B, no mask
            g1[1] = (int)(64u << 16);                         // tensor_dim0=64 (lo16)
            g1[2] = 0;                                        // td0 hi | td1 lo (65536&0xFFFF=0)
            g1[3] = (int)((64u << 16) | 1u);                  // td1 hi=1 | tile_dim0=64
            g1[4] = (int)16u;                                 // tile_dim1=16, tile_dim2=0
            g1[5] = (int)64u;                                 // tensor_dim0_stride=64 (lo32)
            g1[6] = 0;                                        // stride0 hi | stride1 lo
            g1[7] = 0;                                        // stride1 hi
#pragma unroll
            for (int e = 0; e < 4; ++e) { g2[e] = 0; g3[e] = 0; }
#if TDM_SIX_ARG
            v8i g4;
#pragma unroll
            for (int e = 0; e < 8; ++e) g4[e] = 0;
            __builtin_amdgcn_tensor_load_to_lds(g0, g1, g2, g3, g4, 0);
#else
            __builtin_amdgcn_tensor_load_to_lds(g0, g1, g2, g3, 0);
#endif
        }
#else
        {
            __bf16 ktmp[8];
            __builtin_memcpy(ktmp, kp, 16);
            __builtin_memcpy(&Ksh[srow][scol], ktmp, 16);
        }
#endif

#if USE_ASYNC_LDS
        __builtin_amdgcn_global_load_async_to_lds_b128(
            (AS1 v4i_b*)(unsigned long long)(size_t)vp,
            (AS3 v4i_b*)(AS3 void*)&Vraw[srow][scol], 0, 0);
        __builtin_amdgcn_s_wait_asynccnt(0);
#else
        {
            __bf16 vtmp[8];
            __builtin_memcpy(vtmp, vp, 16);
            __builtin_memcpy(&Vraw[srow][scol], vtmp, 16);
        }
#endif
#if USE_TDM
        if (w == 0) __builtin_amdgcn_s_wait_tensorcnt(0);
#endif
        if (j0 + 16 < SEQ) {               // gfx1250 global_prefetch_b8
            __builtin_prefetch(kp + 16 * DK, 0, 1);
            __builtin_prefetch(vp + 16 * DK, 0, 1);
        }
        __syncthreads();                   // K + Vraw visible to all waves

        // transpose V tile: Vraw[16][64] -> VshT[64][16]
        {
            __bf16 t8[8];
            __builtin_memcpy(t8, &Vraw[srow][scol], 16);
#pragma unroll
            for (int e = 0; e < 8; ++e) VshT[scol + e][srow] = t8[e];
        }
        __syncthreads();

        // ---- content scores: C[m,n] = sum_d Q[i0+m,d] * K[j0+n,d] ----
        v16bf bk0 = load16bf(&Ksh[lr][half * 16]);
        v16bf bk1 = load16bf(&Ksh[lr][32 + half * 16]);
        v8f sc = zero8();
        sc = __builtin_amdgcn_wmma_f32_16x16x32_bf16(false, aq0, false, bk0, (short)0, sc, false, false);
        sc = __builtin_amdgcn_wmma_f32_16x16x32_bf16(false, aq1, false, bk1, (short)0, sc, false, false);

        // ---- positional strip: T[m,c] = Q[i0+m] . Rel[roff + c], c in [0,32) ----
        const int roff = j0 - i0 + MAX_LEN - 15;
        v8f t0 = zero8(), t1 = zero8();
        {
            int rr0 = roff + lr;
            int rr1 = roff + 16 + lr;
            rr0 = rr0 < 0 ? 0 : (rr0 > RELROWS - 1 ? RELROWS - 1 : rr0);
            rr1 = rr1 < 0 ? 0 : (rr1 > RELROWS - 1 ? RELROWS - 1 : rr1);
            const __bf16* R0 = Rel + (size_t)rr0 * DK;
            const __bf16* R1 = Rel + (size_t)rr1 * DK;
            v16bf br00 = load16bf(R0 + half * 16);
            v16bf br01 = load16bf(R0 + 32 + half * 16);
            v16bf br10 = load16bf(R1 + half * 16);
            v16bf br11 = load16bf(R1 + 32 + half * 16);
            t0 = __builtin_amdgcn_wmma_f32_16x16x32_bf16(false, aq0, false, br00, (short)0, t0, false, false);
            t0 = __builtin_amdgcn_wmma_f32_16x16x32_bf16(false, aq1, false, br01, (short)0, t0, false, false);
            t1 = __builtin_amdgcn_wmma_f32_16x16x32_bf16(false, aq0, false, br10, (short)0, t1, false, false);
            t1 = __builtin_amdgcn_wmma_f32_16x16x32_bf16(false, aq1, false, br11, (short)0, t1, false, false);
        }
#pragma unroll
        for (int r = 0; r < 8; ++r) {
            Tsh[w][r + 8 * half][lr]      = t0[r];
            Tsh[w][r + 8 * half][16 + lr] = t1[r];
        }
        __builtin_amdgcn_wave_barrier();   // per-wave LDS; DS ops are in-order

        // ---- scores + online softmax (row = 16-lane half-wave group) ----
#pragma unroll
        for (int r = 0; r < 8; ++r) {
            const int m = r + 8 * half;        // tile-local row
            const int c = lr - m + 15;         // shear: pos[m,n] = T[m, n-m+15]
            const float s = (sc[r] + Tsh[w][m][c]) * scale;
            float t = s;
            t = fmaxf(t, __shfl_xor(t, 1, 32));
            t = fmaxf(t, __shfl_xor(t, 2, 32));
            t = fmaxf(t, __shfl_xor(t, 4, 32));
            t = fmaxf(t, __shfl_xor(t, 8, 32));
            const float mnew  = fmaxf(mrow[r], t);
            const float alpha = __expf(mrow[r] - mnew);
            const float pe    = __expf(s - mnew);
            float psum = pe;
            psum += __shfl_xor(psum, 1, 32);
            psum += __shfl_xor(psum, 2, 32);
            psum += __shfl_xor(psum, 4, 32);
            psum += __shfl_xor(psum, 8, 32);
            ssum[r] = ssum[r] * alpha + psum;
            mrow[r] = mnew;
            o0[r] *= alpha; o1[r] *= alpha; o2[r] *= alpha; o3[r] *= alpha;
            Psh[w][m][lr] = f2bf(pe);
        }
        __builtin_amdgcn_wave_barrier();

        // ---- O += P @ V_tile.  A = P padded to K=32 (upper half zero) ----
        v16bf ap;
        __builtin_memcpy(&ap, (const __bf16*)__builtin_assume_aligned(&Psh[w][lr][0], 32), 32);
#pragma unroll
        for (int e = 0; e < 16; ++e) if (half) ap[e] = zb;

        v16bf bv0 = load16bf(&VshT[lr][0]);        // B[k][d]: k&15==e contiguous
        v16bf bv1 = load16bf(&VshT[16 + lr][0]);
        v16bf bv2 = load16bf(&VshT[32 + lr][0]);
        v16bf bv3 = load16bf(&VshT[48 + lr][0]);
        o0 = __builtin_amdgcn_wmma_f32_16x16x32_bf16(false, ap, false, bv0, (short)0, o0, false, false);
        o1 = __builtin_amdgcn_wmma_f32_16x16x32_bf16(false, ap, false, bv1, (short)0, o1, false, false);
        o2 = __builtin_amdgcn_wmma_f32_16x16x32_bf16(false, ap, false, bv2, (short)0, o2, false, false);
        o3 = __builtin_amdgcn_wmma_f32_16x16x32_bf16(false, ap, false, bv3, (short)0, o3, false, false);
    }

    // ---- epilogue: O /= rowsum, write merged-head layout [B*L, D] bf16 ----
    const size_t orow = (size_t)b * SEQ + i0;
#pragma unroll
    for (int r = 0; r < 8; ++r) {
        const int m = r + 8 * half;
        const float inv = 1.0f / ssum[r];
        __bf16* dst = AO + (orow + m) * D_MODEL + h * DK;
        dst[lr]      = f2bf(o0[r] * inv);
        dst[16 + lr] = f2bf(o1[r] * inv);
        dst[32 + lr] = f2bf(o2[r] * inv);
        dst[48 + lr] = f2bf(o3[r] * inv);
    }
}

// ---------------------------------------------------------------------------
// Host-side orchestration
// ---------------------------------------------------------------------------
extern "C" void kernel_launch(void* const* d_in, const int* in_sizes, int n_in,
                              void* d_out, int out_size, void* d_ws, size_t ws_size,
                              hipStream_t stream) {
    const float* q   = (const float*)d_in[0];
    const float* k   = (const float*)d_in[1];
    const float* v   = (const float*)d_in[2];
    const float* rel = (const float*)d_in[3];
    const float* wq  = (const float*)d_in[4];
    const float* bq  = (const float*)d_in[5];
    const float* wk  = (const float*)d_in[6];
    const float* bk  = (const float*)d_in[7];
    const float* wv  = (const float*)d_in[8];
    const float* bv  = (const float*)d_in[9];
    const float* wo  = (const float*)d_in[10];
    const float* bo  = (const float*)d_in[11];
    float* out = (float*)d_out;

    // workspace carve-up (bf16, 256B aligned). Total ~65 MB.
    char* ws = (char*)d_ws;
    size_t off = 0;
    auto alloc = [&](size_t bytes) -> void* {
        void* p = ws + off;
        off += (bytes + 255) & ~(size_t)255;
        return p;
    };
    const size_t actN = (size_t)MTOT * D_MODEL;      // 4096 x 1024
    const size_t wN   = (size_t)D_MODEL * D_MODEL;   // 1024 x 1024
    const size_t relN = (size_t)RELROWS * DK;        // 4096 x 64
    __bf16* qb   = (__bf16*)alloc(actN * 2);
    __bf16* kb   = (__bf16*)alloc(actN * 2);
    __bf16* vb   = (__bf16*)alloc(actN * 2);
    __bf16* wqb  = (__bf16*)alloc(wN * 2);
    __bf16* wkb  = (__bf16*)alloc(wN * 2);
    __bf16* wvb  = (__bf16*)alloc(wN * 2);
    __bf16* wob  = (__bf16*)alloc(wN * 2);
    __bf16* relb = (__bf16*)alloc(relN * 2);
    __bf16* Qhp  = (__bf16*)alloc(actN * 2);         // [B,H,L,dk]
    __bf16* Khp  = (__bf16*)alloc(actN * 2);
    __bf16* Vhp  = (__bf16*)alloc(actN * 2);
    __bf16* AO   = (__bf16*)alloc(actN * 2);         // [B*L, D]
    (void)ws_size; (void)n_in; (void)in_sizes; (void)out_size;

    // 1) fp32 -> bf16 conversions
    const int CT = 256;
    cvt_f32_bf16<<<(int)((actN + CT - 1) / CT), CT, 0, stream>>>(q, qb, (int)actN);
    cvt_f32_bf16<<<(int)((actN + CT - 1) / CT), CT, 0, stream>>>(k, kb, (int)actN);
    cvt_f32_bf16<<<(int)((actN + CT - 1) / CT), CT, 0, stream>>>(v, vb, (int)actN);
    cvt_f32_bf16<<<(int)((wN + CT - 1) / CT),   CT, 0, stream>>>(wq, wqb, (int)wN);
    cvt_f32_bf16<<<(int)((wN + CT - 1) / CT),   CT, 0, stream>>>(wk, wkb, (int)wN);
    cvt_f32_bf16<<<(int)((wN + CT - 1) / CT),   CT, 0, stream>>>(wv, wvb, (int)wN);
    cvt_f32_bf16<<<(int)((wN + CT - 1) / CT),   CT, 0, stream>>>(wo, wob, (int)wN);
    cvt_f32_bf16<<<(int)((relN + CT - 1) / CT), CT, 0, stream>>>(rel, relb, (int)relN);

    // 2) QKV projections (WMMA GEMM, head-split bf16 output)
    dim3 ggrid(MTOT / 16, D_MODEL / 16);
    gemm_wmma_bf16<<<ggrid, 32, 0, stream>>>(qb, wqb, bq, Qhp, nullptr, D_MODEL, 0);
    gemm_wmma_bf16<<<ggrid, 32, 0, stream>>>(kb, wkb, bk, Khp, nullptr, D_MODEL, 0);
    gemm_wmma_bf16<<<ggrid, 32, 0, stream>>>(vb, wvb, bv, Vhp, nullptr, D_MODEL, 0);

    // 3) fused relative attention
    dim3 agrid(SEQ / 64, NUM_HEADS, BATCH);
    rel_attn_wmma<<<agrid, 128, 0, stream>>>(Qhp, Khp, Vhp, relb, AO);

    // 4) output projection, fp32 result
    gemm_wmma_bf16<<<ggrid, 32, 0, stream>>>(AO, wob, bo, nullptr, out, D_MODEL, 1);
}
